// ConvolutionalLayer_59219009077549
// MI455X (gfx1250) — compile-verified, matrix operand order checked by reading, not run on hardware
//
#include <hip/hip_runtime.h>
#include <stdint.h>

typedef __attribute__((ext_vector_type(16))) _Float16 v16h;
typedef __attribute__((ext_vector_type(8)))  _Float16 v8h;
typedef __attribute__((ext_vector_type(4)))  _Float16 v4h;
typedef __attribute__((ext_vector_type(8)))  float    v8f;

#define GCN_N    8192
#define GCN_FIN  1024
#define GCN_FOUT 512

// LDS row pitches (in halfs) padded to vary banks
#define APITCH 40
#define BPITCH 40

// 16-byte async global->LDS copy (ASYNCcnt-tracked), per CDNA5 ISA:
//   GLOBAL_LOAD_ASYNC_TO_LDS_B128  VDST(=LDS byte addr), VADDR(64b), off
// Low 32 bits of a flat pointer to __shared__ data are the LDS byte offset
__device__ inline void async_copy16(const _Float16* __restrict__ g, _Float16* l) {
  asm volatile("global_load_async_to_lds_b128 %0, %1, off"
               :
               : "v"((uint32_t)(uintptr_t)l),
                 "v"((uint64_t)(uintptr_t)g)
               : "memory");
}

__device__ inline void wait_async_and_barrier() {
  asm volatile("s_wait_asynccnt 0x0" ::: "memory");
  __syncthreads();
}

__device__ inline v16h combine16(v8h lo, v8h hi) {
  v16h r;
#pragma unroll
  for (int i = 0; i < 8; ++i) { r[i] = lo[i]; r[i + 8] = hi[i]; }
  return r;
}

__device__ inline v4h cvt4(float4 f) {
  v4h h;
  h[0] = (_Float16)f.x; h[1] = (_Float16)f.y;
  h[2] = (_Float16)f.z; h[3] = (_Float16)f.w;
  return h;
}

// ---------------------------------------------------------------------------
// Kernel 1: d_is[i] = rsqrt(1 + sum_j adj[i][j])   (self loop adds 1)
// ---------------------------------------------------------------------------
__global__ __launch_bounds__(256) void degree_kernel(
    const float* __restrict__ adj, float* __restrict__ dis, int n) {
  __shared__ float red[256];
  const int row = blockIdx.x;
  const float4* a4 = (const float4*)(adj + (size_t)row * n);
  const int n4 = n >> 2;
  float s = 0.0f;
  for (int j = threadIdx.x; j < n4; j += 256) {
    float4 v = a4[j];
    s += v.x + v.y + v.z + v.w;
  }
  red[threadIdx.x] = s;
  __syncthreads();
#pragma unroll
  for (int off = 128; off > 0; off >>= 1) {
    if (threadIdx.x < off) red[threadIdx.x] += red[threadIdx.x + off];
    __syncthreads();
  }
  if (threadIdx.x == 0) dis[row] = rsqrtf(red[0] + 1.0f);
}

// ---------------------------------------------------------------------------
// Kernel 2: Wt[n][k] = (f16) W[k][n]     (WMMA-B-friendly layout, f16)
// ---------------------------------------------------------------------------
__global__ __launch_bounds__(256) void transpose_w_kernel(
    const float* __restrict__ W, _Float16* __restrict__ Wt, int K, int N) {
  int idx = blockIdx.x * 256 + threadIdx.x;
  if (idx >= K * N) return;
  int k = idx / N;
  int n = idx - k * N;
  Wt[(size_t)n * K + k] = (_Float16)W[(size_t)k * N + n];
}

// ---------------------------------------------------------------------------
// Kernel 3: presup = x @ W ; Pt[n][j] = (f16)(dis[j] * presup[j][n])
// Block: 256 thr = 8 waves (2 M x 4 N). Tile: 32 rows x 512 cols.
// Double-buffered LDS; B tiles via async global->LDS; B fragments
// register-double-buffered so ds_load(t+1) overlaps WMMA(t).
// ---------------------------------------------------------------------------
__global__ __launch_bounds__(256) void xw_kernel(
    const float* __restrict__ x,        // [N][FIN] f32
    const _Float16* __restrict__ Wt,    // [FOUT][FIN] f16
    const float* __restrict__ dis,      // [N]
    _Float16* __restrict__ Pt,          // [FOUT][N] f16 (output)
    int n, int fin, int fout) {
  __shared__ _Float16 As[2][32 * APITCH];
  __shared__ _Float16 Bs[2][512 * BPITCH];

  const int tid   = threadIdx.x;
  const int lane  = tid & 31;
  const int wave  = tid >> 5;
  const int waveM = wave & 1;   // 0..1 -> 16 rows each
  const int waveN = wave >> 1;  // 0..3 -> 128 cols each
  const int rowBlock = blockIdx.x * 32;

  const int l16 = lane & 15;
  const int hi  = lane >> 4;
  const int k0  = hi * 8;       // K sub-offset per ISA 16-bit A/B layout

  // staging coordinates
  const int ar = tid >> 3;          // A row 0..31
  const int ac = (tid & 7) * 4;     // A col group 0..28
  const float* aSrc = x + (size_t)(rowBlock + ar) * fin + ac;

  v8f acc[8] = {};

  // ---- prologue: stage tile 0 into buffer 0
  *(v4h*)(&As[0][ar * APITCH + ac]) = cvt4(*(const float4*)(aSrc));
#pragma unroll
  for (int c = 0; c < 8; ++c) {
    int idx = c * 256 + tid;
    int nn  = idx >> 2;
    int ko  = (idx & 3) * 8;
    async_copy16(Wt + (size_t)nn * fin + ko, &Bs[0][nn * BPITCH + ko]);
  }
  wait_async_and_barrier();

  int cur = 0;
  for (int kt = 0; kt < fin; kt += 32) {
    const int nxt = cur ^ 1;
    const bool have_next = (kt + 32) < fin;
    float4 fnext;
    if (have_next) {
      fnext = *(const float4*)(aSrc + kt + 32);         // A f32 for next tile
#pragma unroll
      for (int c = 0; c < 8; ++c) {                     // B async for next tile
        int idx = c * 256 + tid;
        int nn  = idx >> 2;
        int ko  = (idx & 3) * 8;
        async_copy16(Wt + (size_t)nn * fin + (kt + 32) + ko,
                     &Bs[nxt][nn * BPITCH + ko]);
      }
    }

    // ---- compute on current buffer (B fragments pipelined in registers)
    const _Float16* aRow = &As[cur][(waveM * 16 + l16) * APITCH];
    v16h afrag = combine16(*(const v8h*)(aRow + k0),
                           *(const v8h*)(aRow + k0 + 16));
    const _Float16* bBase = &Bs[cur][(waveN * 128 + l16) * BPITCH];
    v16h bfrag = combine16(*(const v8h*)(bBase + k0),
                           *(const v8h*)(bBase + k0 + 16));
#pragma unroll
    for (int t = 0; t < 8; ++t) {
      v16h bnext = bfrag;
      if (t < 7) {
        const _Float16* bCol = bBase + (t + 1) * 16 * BPITCH;
        bnext = combine16(*(const v8h*)(bCol + k0),
                          *(const v8h*)(bCol + k0 + 16));
      }
      acc[t] = __builtin_amdgcn_wmma_f32_16x16x32_f16(
          false, afrag, false, bfrag, (short)0, acc[t], false, false);
      bfrag = bnext;
    }

    if (have_next)
      *(v4h*)(&As[nxt][ar * APITCH + ac]) = cvt4(fnext);

    wait_async_and_barrier();
    cur = nxt;
  }

  // ---- epilogue: Pt[nn][i] = (f16)(dis[i] * acc)
  const int i0 = rowBlock + waveM * 16 + hi * 8;
  float dv[8];
#pragma unroll
  for (int r = 0; r < 8; ++r) dv[r] = dis[i0 + r];
#pragma unroll
  for (int t = 0; t < 8; ++t) {
    int nn = waveN * 128 + t * 16 + l16;
    v8h h;
#pragma unroll
    for (int r = 0; r < 8; ++r) h[r] = (_Float16)(dv[r] * acc[t][r]);
    *(v8h*)(Pt + (size_t)nn * n + i0) = h;   // 8 contiguous halfs along j
  }
}

// ---------------------------------------------------------------------------
// Kernel 4: S[i] = sum_j adj[i][j] * P'[j]  (f16 WMMA, f32 accum)
//           out[i][nn] = relu( dis[i] * (S[i][nn] + P'[i][nn]) + b[nn] )
// Tile: 32 rows x 512 cols (full FOUT -> adj streamed from HBM exactly once;
// Pt re-reads hit L2). Double-buffered LDS; async B copies; pipelined frags.
// ---------------------------------------------------------------------------
__global__ __launch_bounds__(256) void prop_kernel(
    const float* __restrict__ adj,      // [N][N] f32
    const _Float16* __restrict__ Pt,    // [FOUT][N] f16
    const float* __restrict__ dis,      // [N]
    const float* __restrict__ bias,     // [FOUT]
    float* __restrict__ out,            // [N][FOUT] f32
    int n, int fout) {
  __shared__ _Float16 As[2][32 * APITCH];
  __shared__ _Float16 Bs[2][512 * BPITCH];

  const int tid   = threadIdx.x;
  const int lane  = tid & 31;
  const int wave  = tid >> 5;
  const int waveM = wave & 1;
  const int waveN = wave >> 1;
  const int rowBlock = blockIdx.x * 32;

  const int l16 = lane & 15;
  const int hi  = lane >> 4;
  const int k0  = hi * 8;

  const int ar = tid >> 3;
  const int ac = (tid & 7) * 4;
  const float* aSrc = adj + (size_t)(rowBlock + ar) * n + ac;

  v8f acc[8] = {};

  // ---- prologue: stage tile 0 into buffer 0
  *(v4h*)(&As[0][ar * APITCH + ac]) = cvt4(*(const float4*)(aSrc));
#pragma unroll
  for (int c = 0; c < 8; ++c) {
    int idx = c * 256 + tid;
    int nn  = idx >> 2;
    int ko  = (idx & 3) * 8;
    async_copy16(Pt + (size_t)nn * n + ko, &Bs[0][nn * BPITCH + ko]);
  }
  wait_async_and_barrier();

  int cur = 0;
  for (int kt = 0; kt < n; kt += 32) {
    const int nxt = cur ^ 1;
    const bool have_next = (kt + 32) < n;
    float4 fnext;
    if (have_next) {
      fnext = *(const float4*)(aSrc + kt + 32);
#pragma unroll
      for (int c = 0; c < 8; ++c) {
        int idx = c * 256 + tid;
        int nn  = idx >> 2;
        int ko  = (idx & 3) * 8;
        async_copy16(Pt + (size_t)nn * n + (kt + 32) + ko,
                     &Bs[nxt][nn * BPITCH + ko]);
      }
    }

    const _Float16* aRow = &As[cur][(waveM * 16 + l16) * APITCH];
    v16h afrag = combine16(*(const v8h*)(aRow + k0),
                           *(const v8h*)(aRow + k0 + 16));
    const _Float16* bBase = &Bs[cur][(waveN * 128 + l16) * BPITCH];
    v16h bfrag = combine16(*(const v8h*)(bBase + k0),
                           *(const v8h*)(bBase + k0 + 16));
#pragma unroll
    for (int t = 0; t < 8; ++t) {
      v16h bnext = bfrag;
      if (t < 7) {
        const _Float16* bCol = bBase + (t + 1) * 16 * BPITCH;
        bnext = combine16(*(const v8h*)(bCol + k0),
                          *(const v8h*)(bCol + k0 + 16));
      }
      acc[t] = __builtin_amdgcn_wmma_f32_16x16x32_f16(
          false, afrag, false, bfrag, (short)0, acc[t], false, false);
      bfrag = bnext;
    }

    if (have_next)
      *(v4h*)(&As[nxt][ar * APITCH + ac]) = cvt4(fnext);

    wait_async_and_barrier();
    cur = nxt;
  }

  // ---- epilogue: fused self-loop + degree scale + bias + relu
  // Loads/stores arranged so per-row bases are shared and t-steps become
  // immediate offsets (t*16 floats for out, t*16*n halfs for Pt).
  const int i0    = rowBlock + waveM * 16 + hi * 8;
  const int ncol0 = waveN * 128 + l16;
  float dv[8];
#pragma unroll
  for (int r = 0; r < 8; ++r) dv[r] = dis[i0 + r];
  float bv[8];
  v8h   pv[8];
  const _Float16* pBase = Pt + (size_t)ncol0 * n + i0;
#pragma unroll
  for (int t = 0; t < 8; ++t) {
    bv[t] = bias[ncol0 + t * 16];
    pv[t] = *(const v8h*)(pBase + (size_t)t * 16 * n);  // P'[i0..i0+7][ncol0+16t]
  }
#pragma unroll
  for (int r = 0; r < 8; ++r) {
    float* orow = out + (size_t)(i0 + r) * fout + ncol0;
    const float dr = dv[r];
#pragma unroll
    for (int t = 0; t < 8; ++t) {
      float v = dr * (acc[t][r] + (float)pv[t][r]) + bv[t];
      orow[t * 16] = v > 0.0f ? v : 0.0f;
    }
  }
}

// ---------------------------------------------------------------------------
extern "C" void kernel_launch(void* const* d_in, const int* in_sizes, int n_in,
                              void* d_out, int out_size, void* d_ws, size_t ws_size,
                              hipStream_t stream) {
  (void)in_sizes; (void)n_in; (void)out_size; (void)ws_size;
  const float* x   = (const float*)d_in[0];   // [N][FIN]
  const float* adj = (const float*)d_in[1];   // [N][N]
  const float* W   = (const float*)d_in[2];   // [FIN][FOUT]
  const float* b   = (const float*)d_in[3];   // [FOUT]
  float* out = (float*)d_out;                 // [N][FOUT]

  const int N = GCN_N, FIN = GCN_FIN, FOUT = GCN_FOUT;

  char* ws = (char*)d_ws;
  float*    dis = (float*)ws;                                      // 32 KB
  _Float16* Wt  = (_Float16*)(ws + 32768);                         // 1 MB
  _Float16* Pt  = (_Float16*)(ws + 32768 +
                              (size_t)FIN * FOUT * sizeof(_Float16)); // 8 MB

  degree_kernel<<<N, 256, 0, stream>>>(adj, dis, N);
  transpose_w_kernel<<<(FIN * FOUT + 255) / 256, 256, 0, stream>>>(W, Wt, FIN, FOUT);
  xw_kernel<<<N / 32, 256, 0, stream>>>(x, Wt, dis, Pt, N, FIN, FOUT);
  prop_kernel<<<N / 32, 256, 0, stream>>>(adj, Pt, dis, b, out, N, FOUT);
}
